// Connect_Loss_76149770158609
// MI455X (gfx1250) — compile-verified, matrix operand order checked by reading, not run on hardware
//
#include <hip/hip_runtime.h>

// ---------------------------------------------------------------------------
// Connect_Loss for MI455X (gfx1250): morphological opening + normalize +
// binarize + 150-iter masked 3x3 maxpool CCL + distinct-label counting.
// CDNA5 paths used: async global<->LDS copies (ASYNCcnt) for the LDS-tiled
// multi-iteration CCL kernel, and v_wmma_f32_16x16x32_f16 for the boolean
// sum reductions (exact: 0/1 flags in f16, f32 accumulate).
// ---------------------------------------------------------------------------

typedef __attribute__((ext_vector_type(16))) _Float16 v16h;
typedef __attribute__((ext_vector_type(8)))  float    v8f;

#define HH   1024
#define WW   1024
#define BB   4
#define NTOT (BB * HH * WW)   // 4,194,304 = 2^22 (< 2^24, exact in f32)

#define TILE 64
#define KIT  5                 // CCL iterations fused per launch
#define HALO KIT
#define PT   (TILE + 2 * HALO) // 74
#define PST  76                // padded LDS row stride

#define AS1 __attribute__((address_space(1)))
#define AS3 __attribute__((address_space(3)))

#if defined(__has_builtin)
#  if __has_builtin(__builtin_amdgcn_global_load_async_to_lds_b32) && \
      __has_builtin(__builtin_amdgcn_global_load_async_to_lds_b8)  && \
      __has_builtin(__builtin_amdgcn_global_store_async_from_lds_b32)
#    define USE_ASYNC_LDS 1
#  endif
#  if __has_builtin(__builtin_amdgcn_s_wait_asynccnt)
#    define HAVE_WAIT_ASYNC 1
#  endif
#endif

__device__ __forceinline__ void wait_async0() {
#if defined(HAVE_WAIT_ASYNC)
  __builtin_amdgcn_s_wait_asynccnt(0);
#else
  asm volatile("s_wait_asynccnt 0" ::: "memory");
#endif
}

// -------------------------- scalar slot init -------------------------------
// [0]=min bits  [1]=max bits  [2]=compMax count  [3]=sum(mask)
// [4]=ccs  [5]=ccsg  [6]=numSg
__global__ void k_init(unsigned int* sc) {
  sc[0] = 0xFFFFFFFFu;
  sc[1] = 0u;
  ((float*)sc)[2] = 0.0f;
  ((float*)sc)[3] = 0.0f;
}

// -------------------------- erosion (cross, BIG pad) -----------------------
__global__ __launch_bounds__(256) void k_erode(const float* __restrict__ in,
                                               float* __restrict__ out) {
  int idx = blockIdx.x * 256 + threadIdx.x;
  int x = idx & (WW - 1);
  int y = (idx >> 10) & (HH - 1);
  float v = in[idx];
  if (y > 0)      v = fminf(v, in[idx - WW]);
  if (y < HH - 1) v = fminf(v, in[idx + WW]);
  if (x > 0)      v = fminf(v, in[idx - 1]);
  if (x < WW - 1) v = fminf(v, in[idx + 1]);
  out[idx] = v;
}

// ---------------- dilation (cross, -BIG pad) + global min/max --------------
__global__ __launch_bounds__(256) void k_dilate_minmax(const float* __restrict__ in,
                                                       float* __restrict__ out,
                                                       unsigned int* minBits,
                                                       unsigned int* maxBits) {
  int idx = blockIdx.x * 256 + threadIdx.x;
  int x = idx & (WW - 1);
  int y = (idx >> 10) & (HH - 1);
  float v = in[idx];
  if (y > 0)      v = fmaxf(v, in[idx - WW]);
  if (y < HH - 1) v = fmaxf(v, in[idx + WW]);
  if (x > 0)      v = fmaxf(v, in[idx - 1]);
  if (x < WW - 1) v = fmaxf(v, in[idx + 1]);
  out[idx] = v;

  // block reduction (values are in [0,1): nonneg, so u32 bit order == f32 order)
  float mn = v, mx = v;
  for (int o = 16; o > 0; o >>= 1) {
    mn = fminf(mn, __shfl_xor(mn, o, 32));
    mx = fmaxf(mx, __shfl_xor(mx, o, 32));
  }
  __shared__ float smn[8], smx[8];
  int w  = threadIdx.x >> 5;
  int ln = threadIdx.x & 31;
  if (ln == 0) { smn[w] = mn; smx[w] = mx; }
  __syncthreads();
  if (threadIdx.x == 0) {
    for (int i = 1; i < 8; ++i) { mn = fminf(mn, smn[i]); mx = fmaxf(mx, smx[i]); }
    atomicMin(minBits, __float_as_uint(mn));
    atomicMax(maxBits, __float_as_uint(mx));
  }
}

// -------------------- normalize + binarize + label init --------------------
__global__ __launch_bounds__(256) void k_binarize(const float* __restrict__ op,
                                                  const unsigned int* minBits,
                                                  const unsigned int* maxBits,
                                                  unsigned char* __restrict__ mask,
                                                  float* __restrict__ labels) {
  int idx = blockIdx.x * 256 + threadIdx.x;
  float mn = __uint_as_float(*minBits);
  float mx = __uint_as_float(*maxBits);
  float nv = (op[idx] - mn) / (mx - mn + 1e-10f);
  int m = (nv >= 0.5f) ? 1 : 0;
  mask[idx]   = (unsigned char)m;
  labels[idx] = m ? (float)idx : 0.0f;   // exact: idx < 2^22
}

// ---------------- CCL: KIT fused iterations of masked 3x3 maxpool ----------
__global__ __launch_bounds__(256) void k_ccl(const float* __restrict__ in,
                                             float* __restrict__ out,
                                             const unsigned char* __restrict__ mask) {
  __shared__ float t0[PT * PST];
  __shared__ float t1[PT * PST];
  __shared__ unsigned char tm[PT * PST];

  int tid  = threadIdx.x;
  int bx0  = blockIdx.x * TILE - HALO;
  int by0  = blockIdx.y * TILE - HALO;
  int bofs = blockIdx.z * (HH * WW);

  // Load padded tile (labels + mask). Out-of-image cells: label 0, mask 0
  // (safe: all labels >= 0, so an extra 0 never changes a window max).
  for (int i = tid; i < PT * PT; i += 256) {
    int ty = i / PT, tx = i - ty * PT;
    int gy = by0 + ty, gx = bx0 + tx;
    int l  = ty * PST + tx;
    bool inimg = (gy >= 0) && (gy < HH) && (gx >= 0) && (gx < WW);
    if (inimg) {
      int g = bofs + gy * WW + gx;
#ifdef USE_ASYNC_LDS
      __builtin_amdgcn_global_load_async_to_lds_b32(
          (AS1 int*)(void*)(in + g),   (AS3 int*)&t0[l], 0, 0);
      __builtin_amdgcn_global_load_async_to_lds_b8(
          (AS1 char*)(void*)(mask + g), (AS3 char*)&tm[l], 0, 0);
#else
      t0[l] = in[g];
      tm[l] = mask[g];
#endif
    } else {
      t0[l] = 0.0f;
      tm[l] = 0;
    }
  }
#ifdef USE_ASYNC_LDS
  wait_async0();
#endif
  __syncthreads();

  float* cur = t0;
  float* nxt = t1;
  for (int it = 0; it < KIT; ++it) {
    for (int i = tid; i < PT * PT; i += 256) {
      int ty = i / PT, tx = i - ty * PT;
      int l  = ty * PST + tx;
      float c = cur[l];
      float v = c;
      if (tm[l]) {
        if (ty > 0) {
          v = fmaxf(v, cur[l - PST]);
          if (tx > 0)      v = fmaxf(v, cur[l - PST - 1]);
          if (tx < PT - 1) v = fmaxf(v, cur[l - PST + 1]);
        }
        if (ty < PT - 1) {
          v = fmaxf(v, cur[l + PST]);
          if (tx > 0)      v = fmaxf(v, cur[l + PST - 1]);
          if (tx < PT - 1) v = fmaxf(v, cur[l + PST + 1]);
        }
        if (tx > 0)      v = fmaxf(v, cur[l - 1]);
        if (tx < PT - 1) v = fmaxf(v, cur[l + 1]);
      }
      nxt[l] = v;
    }
    __syncthreads();
    float* t = cur; cur = nxt; nxt = t;
  }

  // Write exact interior TILE x TILE.
  for (int i = tid; i < TILE * TILE; i += 256) {
    int ty = i / TILE, tx = i - ty * TILE;
    int gy = blockIdx.y * TILE + ty;
    int gx = blockIdx.x * TILE + tx;
    int g  = bofs + gy * WW + gx;
    int l  = (ty + HALO) * PST + (tx + HALO);
#ifdef USE_ASYNC_LDS
    __builtin_amdgcn_global_store_async_from_lds_b32(
        (AS1 int*)(out + g), (AS3 int*)&cur[l], 0, 0);
#else
    out[g] = cur[l];
#endif
  }
#ifdef USE_ASYNC_LDS
  wait_async0();
#endif
}

// ------- WMMA-based reductions: #component-maxes and sum(mask) -------------
// A = 16x32 f16 flag matrix (per wave: 32 lanes x 16 flags = 512 elements),
// B = ones, C accumulates in f32 -> column sum of D = total count (exact).
__global__ __launch_bounds__(256) void k_count(const float* __restrict__ labels,
                                               const unsigned char* __restrict__ mask,
                                               float* cnt, float* summ) {
  int tid  = threadIdx.x;
  int lane = tid & 31;
  int gw   = (blockIdx.x * 256 + tid) >> 5;     // global wave id
  int nw   = gridDim.x * 8;                     // total waves

  v8f accC = {0.f, 0.f, 0.f, 0.f, 0.f, 0.f, 0.f, 0.f};
  v8f accM = {0.f, 0.f, 0.f, 0.f, 0.f, 0.f, 0.f, 0.f};
  v16h ones;
#pragma unroll
  for (int j = 0; j < 16; ++j) ones[j] = (_Float16)1.0f;

  for (long base = (long)gw * 512; base + 512 <= (long)NTOT; base += (long)nw * 512) {
    int off = (int)base + lane * 16;
    v16h fc, fm;
#pragma unroll
    for (int j = 0; j < 16; ++j) {
      int idx = off + j;
      unsigned char mk = mask[idx];
      float lb = labels[idx];
      fm[j] = mk ? (_Float16)1.0f : (_Float16)0.0f;
      fc[j] = (mk && (lb == (float)idx)) ? (_Float16)1.0f : (_Float16)0.0f;
    }
    accC = __builtin_amdgcn_wmma_f32_16x16x32_f16(false, fc, false, ones,
                                                  (short)0, accC, false, false);
    accM = __builtin_amdgcn_wmma_f32_16x16x32_f16(false, fm, false, ones,
                                                  (short)0, accM, false, false);
  }

  // One column of D: lanes ln and ln+16 jointly hold rows 0..15.
  float sC = 0.f, sM = 0.f;
#pragma unroll
  for (int j = 0; j < 8; ++j) { sC += accC[j]; sM += accM[j]; }
  sC += __shfl_xor(sC, 16, 32);
  sM += __shfl_xor(sM, 16, 32);
  if (lane == 0) {
    atomicAdd(cnt,  sC);   // integer-valued f32 adds: exact & order-independent
    atomicAdd(summ, sM);
  }
}

// ----------------- distinct-count fixup + stash results --------------------
__global__ void k_combine(const float* __restrict__ labels,
                          const unsigned char* __restrict__ mask,
                          const float* cnt, const float* summ,
                          float* distinctOut, float* sumOut) {
  float c = *cnt;
  float s = *summ;
  bool zeroIsMax = (mask[0] != 0) && (labels[0] == 0.0f);
  float d = c + (((s < (float)NTOT) && !zeroIsMax) ? 1.0f : 0.0f);
  *distinctOut = d;
  if (sumOut) *sumOut = s;
}

__global__ void k_final(const float* sc, float* out) {
  float ccs = sc[4], ccsg = sc[5], numSg = sc[6];
  out[0] = fabsf(ccsg - ccs) / numSg;
}

// ---------------------------------------------------------------------------
extern "C" void kernel_launch(void* const* d_in, const int* in_sizes, int n_in,
                              void* d_out, int out_size, void* d_ws, size_t ws_size,
                              hipStream_t stream) {
  (void)in_sizes; (void)n_in; (void)out_size; (void)ws_size;

  const float* img = (const float*)d_in[0];
  const float* lab = (const float*)d_in[1];
  float* out = (float*)d_out;

  char* ws = (char*)d_ws;
  float*         bufA  = (float*)ws;
  float*         bufB  = (float*)(ws + (size_t)16 * 1024 * 1024);
  unsigned char* mk    = (unsigned char*)(ws + (size_t)32 * 1024 * 1024);
  unsigned int*  scalU = (unsigned int*)(ws + (size_t)36 * 1024 * 1024);
  float*         scalF = (float*)scalU;

  dim3 blk(256);
  dim3 grd1(NTOT / 256);
  dim3 grdT(WW / TILE, HH / TILE, BB);

  const float* srcs[2] = {img, lab};
  for (int p = 0; p < 2; ++p) {
    k_init<<<1, 1, 0, stream>>>(scalU);
    k_erode<<<grd1, blk, 0, stream>>>(srcs[p], bufA);
    k_dilate_minmax<<<grd1, blk, 0, stream>>>(bufA, bufB, &scalU[0], &scalU[1]);
    k_binarize<<<grd1, blk, 0, stream>>>(bufB, &scalU[0], &scalU[1], mk, bufA);

    float* a = bufA;
    float* b = bufB;
    for (int it = 0; it < 150 / KIT; ++it) {   // 30 launches x 5 iters = 150
      k_ccl<<<grdT, blk, 0, stream>>>(a, b, mk);
      float* t = a; a = b; b = t;
    }
    // 30 swaps (even) -> final labels back in bufA (== a)
    k_count<<<256, blk, 0, stream>>>(a, mk, &scalF[2], &scalF[3]);
    k_combine<<<1, 1, 0, stream>>>(a, mk, &scalF[2], &scalF[3],
                                   (p == 0) ? &scalF[4] : &scalF[5],
                                   (p == 0) ? (float*)nullptr : &scalF[6]);
  }
  k_final<<<1, 1, 0, stream>>>(scalF, out);
}
